// SpiralMeshReader_65824668779070
// MI455X (gfx1250) — matrix operand order repaired and checked by reading, not compiled, not run on hardware
//
#include <hip/hip_runtime.h>

typedef __attribute__((ext_vector_type(16))) _Float16 v16h;
typedef __attribute__((ext_vector_type(8)))  float    v8f;

#define NN 50000
#define NE 600000
#define EPS 1e-5f
#define SLOPE 0.01f

// ---------------------------------------------------------------------------
// Degree accumulation (unweighted out-deg of src, in-deg of dst)
// ---------------------------------------------------------------------------
__global__ void k_degree(const int* __restrict__ src, const int* __restrict__ dst,
                         float* __restrict__ outdeg, float* __restrict__ indeg, int E) {
  int i = blockIdx.x * blockDim.x + threadIdx.x;
  int stride = gridDim.x * blockDim.x;
  for (; i < E; i += stride) {
    if (i + stride < E) {               // hide index-fetch latency of next chunk
      __builtin_prefetch(&src[i + stride], 0, 0);   // -> global_prefetch_b8
      __builtin_prefetch(&dst[i + stride], 0, 0);
    }
    atomicAdd(&outdeg[src[i]], 1.0f);
    atomicAdd(&indeg[dst[i]], 1.0f);
  }
}

// deg -> rsqrt(max(deg,1)) in place (operates on outdeg||indeg contiguously)
__global__ void k_rsqrt_deg(float* __restrict__ deg, int n) {
  int i = blockIdx.x * blockDim.x + threadIdx.x;
  if (i < n) {
    float v = deg[i];
    v = v < 1.0f ? 1.0f : v;
    deg[i] = rsqrtf(v);
  }
}

// ---------------------------------------------------------------------------
// WMMA GEMM: Y[M,NOUT] = (X[M,K] @ W[K,NOUT]) * rowScale[m]
// One block = one 16-row M-tile, NOUT/16 waves, wave w owns N-tile w.
// A (16x32) staged once per K-step, shared by all waves; B (32xNOUT) staged
// per K-step. All global->LDS staging uses float4 (global_load_b128).
// Fragment packing per CDNA5 ISA 7.12.2 (16-bit A 16x32, B 32x16, f32 C/D).
// ---------------------------------------------------------------------------
template<int K, int NOUT>
__global__ __launch_bounds__(32 * (NOUT / 16))
void k_gemm_wmma(const float* __restrict__ X, const float* __restrict__ W,
                 const float* __restrict__ rowScale, float* __restrict__ Y) {
  constexpr int NW   = NOUT / 16;     // waves per block
  constexpr int NTHR = 32 * NW;
  __shared__ _Float16 As[16 * 32];    // [m][kk]
  __shared__ _Float16 Bs[32 * NOUT];  // [kk][n]

  const int lane = threadIdx.x;       // 0..31 (wave32)
  const int wv   = threadIdx.y;       // 0..NW-1
  const int tid  = wv * 32 + lane;
  const int m0   = blockIdx.x * 16;
  const int n0   = wv * 16;

  v8f acc = {};
  for (int k0 = 0; k0 < K; k0 += 32) {
    // stage A tile: 16 rows x 32 cols = 128 float4 vectors
    #pragma unroll
    for (int i = tid; i < 16 * 8; i += NTHR) {
      int m = i >> 3, vk = i & 7;
      float4 v = reinterpret_cast<const float4*>(X + (size_t)(m0 + m) * K + k0)[vk];
      int o = m * 32 + vk * 4;
      As[o + 0] = (_Float16)v.x; As[o + 1] = (_Float16)v.y;
      As[o + 2] = (_Float16)v.z; As[o + 3] = (_Float16)v.w;
    }
    // stage B tile: 32 rows x NOUT cols = 32*(NOUT/4) float4 vectors
    #pragma unroll
    for (int i = tid; i < 32 * (NOUT / 4); i += NTHR) {
      int kk = i / (NOUT / 4), vn = i % (NOUT / 4);
      float4 v = reinterpret_cast<const float4*>(W + (size_t)(k0 + kk) * NOUT)[vn];
      int o = kk * NOUT + vn * 4;
      Bs[o + 0] = (_Float16)v.x; Bs[o + 1] = (_Float16)v.y;
      Bs[o + 2] = (_Float16)v.z; Bs[o + 3] = (_Float16)v.w;
    }
    __syncthreads();

    const int half = lane >> 4;
    const int mr   = lane & 15;
    v16h a, b;
    #pragma unroll
    for (int j = 0; j < 16; ++j) {
      int v = j >> 1, sub = j & 1;
      int kl = (v < 4) ? (8 * half + 2 * v + sub)
                       : (16 + 8 * half + 2 * (v - 4) + sub);
      a[j] = As[mr * 32 + kl];            // contiguous runs -> ds_load_b128
      b[j] = Bs[kl * NOUT + n0 + mr];     // kl = K row, lane%16 = column
    }
    acc = __builtin_amdgcn_wmma_f32_16x16x32_f16(
        /*neg_a=*/false, a, /*neg_b=*/false, b,
        /*c_mod=*/(short)0, acc, /*reuse_a=*/false, /*reuse_b=*/false);
    __syncthreads();
  }

  // D layout: VGPR r -> row r + 8*(lane/16); lane%16 -> column
  const int half = lane >> 4;
  const int nn   = lane & 15;
  #pragma unroll
  for (int r = 0; r < 8; ++r) {
    int row = m0 + r + 8 * half;
    Y[(size_t)row * NOUT + n0 + nn] = acc[r] * rowScale[row];
  }
}

// ---------------------------------------------------------------------------
// Edge scatter: agg[dst] += h[src] * ew   (one block handles 256/C edges,
// all C channels of an edge contiguous -> coalesced gather + atomics)
// ---------------------------------------------------------------------------
template<int C>
__global__ void k_scatter(const float* __restrict__ h, const int* __restrict__ src,
                          const int* __restrict__ dst, const float* __restrict__ ew,
                          float* __restrict__ agg, int E) {
  const int epb = 256 / C;
  int e = blockIdx.x * epb + threadIdx.x / C;
  int c = threadIdx.x & (C - 1);
  if (e < E) {
    int s = src[e], d = dst[e];
    float w = ew[e];
    atomicAdd(&agg[(size_t)d * C + c], h[(size_t)s * C + c] * w);
  }
}

// out = leaky_relu(agg * in_norm[n])  (float4 vectorized; C % 4 == 0)
template<int C>
__global__ void k_innorm_leaky(const float4* __restrict__ agg, const float* __restrict__ ininv,
                               float4* __restrict__ out, int M) {
  int i = blockIdx.x * blockDim.x + threadIdx.x;   // float4 index
  if (i < M * (C / 4)) {
    int n = i / (C / 4);
    float s = ininv[n];
    float4 g = agg[i];
    float4 r;
    float v;
    v = g.x * s; r.x = v > 0.0f ? v : SLOPE * v;
    v = g.y * s; r.y = v > 0.0f ? v : SLOPE * v;
    v = g.z * s; r.z = v > 0.0f ? v : SLOPE * v;
    v = g.w * s; r.w = v > 0.0f ? v : SLOPE * v;
    out[i] = r;
  }
}

// ---------------------------------------------------------------------------
// Per-channel column reductions for GraphNorm (LDS reduce, 1 atomic/channel)
// ---------------------------------------------------------------------------
template<int C>
__global__ void k_colsum(const float* __restrict__ h, float* __restrict__ sum, int M) {
  __shared__ float red[8][32];
  int c = blockIdx.y * 32 + threadIdx.x;
  float p = 0.0f;
  for (int n = blockIdx.x * 8 + threadIdx.y; n < M; n += gridDim.x * 8)
    p += h[(size_t)n * C + c];
  red[threadIdx.y][threadIdx.x] = p;
  __syncthreads();
  if (threadIdx.y == 0) {
    float t = 0.0f;
    #pragma unroll
    for (int r = 0; r < 8; ++r) t += red[r][threadIdx.x];
    atomicAdd(&sum[c], t);
  }
}

template<int C>
__global__ void k_colvar(const float* __restrict__ h, const float* __restrict__ alpha,
                         const float* __restrict__ mean, float* __restrict__ sq, int M) {
  __shared__ float red[8][32];
  int c = blockIdx.y * 32 + threadIdx.x;
  float am = alpha[c] * mean[c];
  float p = 0.0f;
  for (int n = blockIdx.x * 8 + threadIdx.y; n < M; n += gridDim.x * 8) {
    float s = h[(size_t)n * C + c] - am;
    p += s * s;
  }
  red[threadIdx.y][threadIdx.x] = p;
  __syncthreads();
  if (threadIdx.y == 0) {
    float t = 0.0f;
    #pragma unroll
    for (int r = 0; r < 8; ++r) t += red[r][threadIdx.x];
    atomicAdd(&sq[c], t);
  }
}

__global__ void k_finalize_mean(float* __restrict__ s, int C) {
  int c = threadIdx.x;
  if (c < C) s[c] *= (1.0f / (float)NN);
}
__global__ void k_finalize_istd(float* __restrict__ s, int C) {
  int c = threadIdx.x;
  if (c < C) s[c] = rsqrtf(s[c] * (1.0f / (float)NN) + EPS);
}

// h = gamma*(h - alpha*mean)*istd + beta   (float4 vectorized)
template<int C>
__global__ void k_gnorm(float4* __restrict__ h, const float* __restrict__ gamma,
                        const float* __restrict__ beta, const float* __restrict__ alpha,
                        const float* __restrict__ mean, const float* __restrict__ istd, int M) {
  int i = blockIdx.x * blockDim.x + threadIdx.x;   // float4 index
  if (i < M * (C / 4)) {
    int c0 = (i * 4) & (C - 1);
    float4 v = h[i];
    float4 r;
    r.x = gamma[c0 + 0] * (v.x - alpha[c0 + 0] * mean[c0 + 0]) * istd[c0 + 0] + beta[c0 + 0];
    r.y = gamma[c0 + 1] * (v.y - alpha[c0 + 1] * mean[c0 + 1]) * istd[c0 + 1] + beta[c0 + 1];
    r.z = gamma[c0 + 2] * (v.z - alpha[c0 + 2] * mean[c0 + 2]) * istd[c0 + 2] + beta[c0 + 2];
    r.w = gamma[c0 + 3] * (v.w - alpha[c0 + 3] * mean[c0 + 3]) * istd[c0 + 3] + beta[c0 + 3];
    h[i] = r;
  }
}

// ---------------------------------------------------------------------------
// Readout: per-channel mean/max/min over nodes. Floats mapped to order-
// preserving u32 keys so integer atomicMax/Min work.
// ---------------------------------------------------------------------------
__device__ __forceinline__ unsigned fkey(float f) {
  unsigned u = __float_as_uint(f);
  return (u >> 31) ? ~u : (u | 0x80000000u);
}
__device__ __forceinline__ float unfkey(unsigned k) {
  return (k >> 31) ? __uint_as_float(k & 0x7fffffffu) : __uint_as_float(~k);
}

__global__ void k_readout(const float* __restrict__ h, float* __restrict__ rsum,
                          unsigned* __restrict__ rmax, unsigned* __restrict__ rmin, int M) {
  __shared__ float    ssum[4][64];
  __shared__ unsigned smax[4][64];
  __shared__ unsigned smin[4][64];
  int c  = threadIdx.x;   // 0..63 channel
  int ty = threadIdx.y;   // 0..3
  float ps = 0.0f, pmx = -__builtin_inff(), pmn = __builtin_inff();
  for (int n = blockIdx.x * 4 + ty; n < M; n += gridDim.x * 4) {
    float v = h[(size_t)n * 64 + c];
    ps += v;
    pmx = fmaxf(pmx, v);
    pmn = fminf(pmn, v);
  }
  ssum[ty][c] = ps; smax[ty][c] = fkey(pmx); smin[ty][c] = fkey(pmn);
  __syncthreads();
  if (ty == 0) {
    float s = ssum[0][c];
    unsigned mx = smax[0][c], mn = smin[0][c];
    #pragma unroll
    for (int r = 1; r < 4; ++r) {
      s += ssum[r][c];
      mx = max(mx, smax[r][c]);
      mn = min(mn, smin[r][c]);
    }
    atomicAdd(&rsum[c], s);
    atomicMax(&rmax[c], mx);
    atomicMin(&rmin[c], mn);
  }
}

__global__ void k_finalize_readout(const float* __restrict__ rsum,
                                   const unsigned* __restrict__ rmax,
                                   const unsigned* __restrict__ rmin,
                                   float* __restrict__ r) {
  int c = threadIdx.x;
  if (c < 64) {
    r[c]       = rsum[c] * (1.0f / (float)NN);
    r[64 + c]  = unfkey(rmax[c]);
    r[128 + c] = unfkey(rmin[c]);
  }
}

// out[1,32] = r[1,192] @ Wc[192,32]
__global__ void k_final_gemm(const float* __restrict__ r, const float* __restrict__ Wc,
                             float* __restrict__ out) {
  int j = threadIdx.x; // 0..31
  float acc = 0.0f;
  #pragma unroll 4
  for (int k = 0; k < 192; ++k) acc += r[k] * Wc[k * 32 + j];
  out[j] = acc;
}

// ---------------------------------------------------------------------------
extern "C" void kernel_launch(void* const* d_in, const int* in_sizes, int n_in,
                              void* d_out, int out_size, void* d_ws, size_t ws_size,
                              hipStream_t stream) {
  const float* x   = (const float*)d_in[0];
  const int*   src = (const int*)d_in[1];
  const int*   dst = (const int*)d_in[2];
  const float* ew  = (const float*)d_in[3];
  const float* W1  = (const float*)d_in[4];   // [64,128]
  const float* W2  = (const float*)d_in[5];   // [128,64]
  const float* g1  = (const float*)d_in[6];
  const float* b1  = (const float*)d_in[7];
  const float* a1  = (const float*)d_in[8];
  const float* g2  = (const float*)d_in[9];
  const float* b2  = (const float*)d_in[10];
  const float* a2  = (const float*)d_in[11];
  const float* Wc  = (const float*)d_in[12];  // [192,32]
  float* out = (float*)d_out;

  const int N = NN, E = NE;

  // workspace layout (floats)
  float* f      = (float*)d_ws;
  float* outdeg = f;                           // N  -> becomes out-norm (rsqrt)
  float* indeg  = f + N;                       // N  -> becomes in-norm (rsqrt)
  float* h1     = f + 2 * (size_t)N;           // N*128
  float* aggA   = h1 + (size_t)N * 128;        // N*128 : agg1, later h2|agg2
  float* h2     = aggA;                        // N*64  (aliases agg1 lower half)
  float* agg2   = aggA + (size_t)N * 64;       // N*64  (aliases agg1 upper half)
  float* stats  = aggA + (size_t)N * 128;
  float*    sum1 = stats;                      // 128 -> mean1 (in place)
  float*    sq1  = stats + 128;                // 128 -> istd1 (in place)
  float*    sum2 = stats + 256;                // 64  -> mean2
  float*    sq2  = stats + 320;                // 64  -> istd2
  float*    rsum = stats + 384;                // 64
  unsigned* rmax = (unsigned*)(stats + 448);   // 64
  unsigned* rmin = (unsigned*)(stats + 512);   // 64
  float*    rvec = stats + 576;                // 192

  // zero init (graph-capture-legal stream memsets)
  hipMemsetAsync(outdeg, 0, sizeof(float) * 2 * (size_t)N, stream);
  hipMemsetAsync(aggA,   0, sizeof(float) * (size_t)N * 128, stream);
  hipMemsetAsync(stats,  0, sizeof(float) * 768, stream);
  hipMemsetAsync(rmin, 0xFF, sizeof(unsigned) * 64, stream);   // +inf key

  // degrees -> rsqrt norms
  k_degree<<<1024, 256, 0, stream>>>(src, dst, outdeg, indeg, E);
  k_rsqrt_deg<<<(2 * N + 255) / 256, 256, 0, stream>>>(outdeg, 2 * N);

  // ---- layer 1: conv(64->128) ----
  k_gemm_wmma<64, 128><<<N / 16, dim3(32, 8), 0, stream>>>(x, W1, outdeg, h1);
  k_scatter<128><<<E / 2, 256, 0, stream>>>(h1, src, dst, ew, aggA, E);
  k_innorm_leaky<128><<<(N * 32 + 255) / 256, 256, 0, stream>>>(
      (const float4*)aggA, indeg, (float4*)h1, N);
  k_colsum<128><<<dim3(64, 4), dim3(32, 8), 0, stream>>>(h1, sum1, N);
  k_finalize_mean<<<1, 128, 0, stream>>>(sum1, 128);
  k_colvar<128><<<dim3(64, 4), dim3(32, 8), 0, stream>>>(h1, a1, sum1, sq1, N);
  k_finalize_istd<<<1, 128, 0, stream>>>(sq1, 128);
  k_gnorm<128><<<(N * 32 + 255) / 256, 256, 0, stream>>>(
      (float4*)h1, g1, b1, a1, sum1, sq1, N);

  // ---- layer 2: conv(128->64) ----
  k_gemm_wmma<128, 64><<<N / 16, dim3(32, 4), 0, stream>>>(h1, W2, outdeg, h2);
  hipMemsetAsync(agg2, 0, sizeof(float) * (size_t)N * 64, stream);
  k_scatter<64><<<E / 4, 256, 0, stream>>>(h2, src, dst, ew, agg2, E);
  k_innorm_leaky<64><<<(N * 16 + 255) / 256, 256, 0, stream>>>(
      (const float4*)agg2, indeg, (float4*)agg2, N);
  k_colsum<64><<<dim3(64, 2), dim3(32, 8), 0, stream>>>(agg2, sum2, N);
  k_finalize_mean<<<1, 64, 0, stream>>>(sum2, 64);
  k_colvar<64><<<dim3(64, 2), dim3(32, 8), 0, stream>>>(agg2, a2, sum2, sq2, N);
  k_finalize_istd<<<1, 64, 0, stream>>>(sq2, 64);
  k_gnorm<64><<<(N * 16 + 255) / 256, 256, 0, stream>>>(
      (float4*)agg2, g2, b2, a2, sum2, sq2, N);

  // ---- readout + final projection ----
  k_readout<<<128, dim3(64, 4), 0, stream>>>(agg2, rsum, rmax, rmin, N);
  k_finalize_readout<<<1, 64, 0, stream>>>(rsum, rmax, rmin, rvec);
  k_final_gemm<<<1, 32, 0, stream>>>(rvec, Wc, out);
}
